// Transformer_Model2_50680614093531
// MI455X (gfx1250) — matrix operand order repaired and checked by reading, not compiled, not run on hardware
//
#include <hip/hip_runtime.h>
#include <hip/hip_bf16.h>

typedef __attribute__((ext_vector_type(16))) _Float16 v16h;
typedef __attribute__((ext_vector_type(8)))  _Float16 v8h;
typedef __attribute__((ext_vector_type(8)))  float    v8f;
typedef __attribute__((ext_vector_type(4)))  int      v4i;

#define B_  4
#define S_  1024
#define D_  1024
#define H_  16
#define L_  8
#define V_  32000
#define DL_ 512
#define HD_ 64
#define M_  (B_*S_)

#define PADH 16

// gfx1250 async global->LDS path (ASYNCcnt-tracked). Toolchain signature:
//   void __builtin_amdgcn_global_load_async_to_lds_b128(v4i AS1* src, v4i AS3* dst,
//                                                       imm int offset, imm int cpol)
#ifndef TRY_ASYNC_LDS
#define TRY_ASYNC_LDS 1
#endif
#if TRY_ASYNC_LDS && defined(__has_builtin)
#if __has_builtin(__builtin_amdgcn_global_load_async_to_lds_b128)
#define USE_ASYNC_LDS 1
#endif
#endif

typedef __attribute__((address_space(1))) v4i* as1_v4i_p;
typedef __attribute__((address_space(3))) v4i* as3_v4i_p;

// ---------------------------------------------------------------- helpers
__device__ __forceinline__ v8h ld8(const _Float16* p) { return *(const v8h*)p; }

__device__ __forceinline__ v16h cat8(v8h lo, v8h hi) {
  v16h r;
#pragma unroll
  for (int i = 0; i < 8; ++i) { r[i] = lo[i]; r[8 + i] = hi[i]; }
  return r;
}

__device__ __forceinline__ v8f wmma_f16(v16h a, v16h b, v8f c) {
  return __builtin_amdgcn_wmma_f32_16x16x32_f16(false, a, false, b, (short)0, c,
                                                false, false);
}

__device__ __forceinline__ void async_copy16(const _Float16* gp, _Float16* lp) {
#if USE_ASYNC_LDS
  __builtin_amdgcn_global_load_async_to_lds_b128((as1_v4i_p)gp, (as3_v4i_p)lp, 0, 0);
#else
  *(v8h*)lp = ld8(gp);
#endif
}

__device__ __forceinline__ void wait_async_lds() {
#if USE_ASYNC_LDS
#if defined(__has_builtin) && __has_builtin(__builtin_amdgcn_s_wait_asynccnt)
  __builtin_amdgcn_s_wait_asynccnt(0);
#else
  asm volatile("s_wait_asynccnt 0" ::: "memory");
#endif
#endif
}

// ---------------------------------------------------------------- fp32 -> f16 (transposed)
// out[l][n][k] = (f16) in[l][k][n];  K = 1<<kshift, N = perMat>>kshift
__global__ void cvtT(const float* __restrict__ in, _Float16* __restrict__ out,
                     int kshift, int perMat) {
  int l = blockIdx.y;
  int i = blockIdx.x * 256 + threadIdx.x;
  if (i >= perMat) return;
  int K = 1 << kshift;
  int n = i >> kshift;
  int k = i & (K - 1);
  int N = perMat >> kshift;
  size_t base = (size_t)l * perMat;
  out[base + i] = (_Float16)in[base + (size_t)k * N + n];
}

// ---------------------------------------------------------------- embedding + PE
__global__ void embed_pe(const int* __restrict__ src, const float* __restrict__ emb,
                         _Float16* __restrict__ x) {
  int idx = blockIdx.x * 256 + threadIdx.x;   // over M_*D_
  int d  = idx & (D_ - 1);
  int bs = idx >> 10;                          // / D_
  int s  = bs & (S_ - 1);
  int tok = src[bs];
  tok = tok < 0 ? 0 : (tok >= V_ ? V_ - 1 : tok);
  int i2 = d & ~1;
  float div = __expf(-9.210340371976184f * (float)i2 / (float)D_); // ln(10000)
  float ang = (float)s * div;
  float pe = (d & 1) ? __cosf(ang) : __sinf(ang);
  x[idx] = (_Float16)(emb[(size_t)tok * D_ + d] + pe);
}

// ---------------------------------------------------------------- WMMA GEMM
// C[M,N] = (A (+A2)) [M,K] * B [K,N]  with B supplied TRANSPOSED as Bt[N][K].
// Block: 256 thr = 8 waves; block tile 128x128; wave tile 32x64; K step 64.
// A staged in LDS via async loads; B-fragments streamed straight from L2
// (each lane's fragment is two 16B runs of Bt, 32B apart, thanks to the pre-transpose).
template <bool FUSED, bool F32OUT>
__global__ __launch_bounds__(256) void gemm2(
    const _Float16* __restrict__ A, const _Float16* __restrict__ A2,
    const _Float16* __restrict__ Bt, const float* __restrict__ bias,
    _Float16* __restrict__ outH, float* __restrict__ outF,
    int M, int N, int K) {
  __shared__ alignas(32) _Float16 sA[128][64 + PADH];

  const int tid = threadIdx.x;
  const int lane = tid & 31, wave = tid >> 5;
  const int hi = lane >> 4, ln = lane & 15;
  const int wm = (wave >> 1) * 32;   // 0,32,64,96
  const int wn = (wave & 1) * 64;    // 0,64
  const int bm = blockIdx.y * 128, bn = blockIdx.x * 128;

  v8f acc[2][4] = {};

  // per-wave B fragment base pointers: column n = bn+wn+j*16+ln, K-chunks hi*8 and 16+hi*8
  const _Float16* brow[4];
#pragma unroll
  for (int j = 0; j < 4; ++j)
    brow[j] = Bt + (size_t)(bn + wn + j * 16 + ln) * K + hi * 8;

  for (int k0 = 0; k0 < K; k0 += 64) {
    // ---- stage A: 128x64 halves = 1024 x 16B chunks, 4 per thread
#pragma unroll
    for (int it = 0; it < 4; ++it) {
      int c = tid + it * 256;
      int row = c >> 3;
      int col = (c & 7) * 8;
      const _Float16* gp = &A[(size_t)(bm + row) * K + k0 + col];
      if (FUSED) {
        v8h va = ld8(gp) + ld8(&A2[(size_t)(bm + row) * K + k0 + col]);
        *(v8h*)&sA[row][col] = va;
      } else {
        async_copy16(gp, &sA[row][col]);
      }
    }
    if (k0 + 64 < K)  // gfx1250 global_prefetch_b8 of next A tile
      __builtin_prefetch(&A[(size_t)(bm + (tid >> 1)) * K + k0 + 64], 0, 0);
    if (!FUSED) wait_async_lds();
    __syncthreads();

    // ---- compute: 2 k-steps x (2 m-tiles x 4 n-tiles) = 16 WMMAs
#pragma unroll
    for (int ks = 0; ks < 64; ks += 32) {
      v16h a0 = cat8(ld8(&sA[wm + ln][ks + hi * 8]),
                     ld8(&sA[wm + ln][ks + 16 + hi * 8]));
      v16h a1 = cat8(ld8(&sA[wm + 16 + ln][ks + hi * 8]),
                     ld8(&sA[wm + 16 + ln][ks + 16 + hi * 8]));
#pragma unroll
      for (int j = 0; j < 4; ++j) {
        const _Float16* bp = brow[j] + k0 + ks;
        v16h b = cat8(ld8(bp), ld8(bp + 16));
        acc[0][j] = wmma_f16(a0, b, acc[0][j]);
        acc[1][j] = wmma_f16(a1, b, acc[1][j]);
      }
    }
    __syncthreads();
  }

  // ---- epilogue: C frag -> row m = r + 8*hi, col n = ln
#pragma unroll
  for (int i = 0; i < 2; ++i) {
#pragma unroll
    for (int j = 0; j < 4; ++j) {
      int rowbase = bm + wm + i * 16 + 8 * hi;
      int col = bn + wn + j * 16 + ln;
      float bv = F32OUT ? bias[col] : 0.0f;
#pragma unroll
      for (int r = 0; r < 8; ++r) {
        float val = acc[i][j][r] + bv;
        size_t idx = (size_t)(rowbase + r) * N + col;
        if (F32OUT) outF[idx] = val;
        else        outH[idx] = (_Float16)val;
      }
    }
  }
}

// ---------------------------------------------------------------- attention
// Flash-style: block = 4 waves (128 thr), each wave 16 query rows; key chunks of 32.
// scores = (Q Kr^T) * scale * exp(-0.01|i-j|); online softmax; O = P Vr.
__global__ __launch_bounds__(128) void attn_wmma(
    const _Float16* __restrict__ Q, const _Float16* __restrict__ Kx,
    const _Float16* __restrict__ V, _Float16* __restrict__ O, float scale) {
  __shared__ alignas(32) _Float16 sK[32][HD_ + 16];   // [key][dim]
  __shared__ alignas(32) _Float16 sVt[HD_][32 + 16];  // [dim][key]
  __shared__ alignas(32) _Float16 sP[4][16][32];      // per-wave probs staging

  const int tid = threadIdx.x;
  const int lane = tid & 31, wave = tid >> 5;
  const int hi = lane >> 4, ln = lane & 15;
  const int bh = blockIdx.y;
  const int b = bh / H_, h = bh % H_;
  const int q0 = blockIdx.x * 64 + wave * 16;

  const _Float16* qptr = Q + (size_t)b * S_ * D_ + (size_t)h * HD_;
  const _Float16* kptr = Kx + (size_t)b * S_ * D_ + (size_t)h * HD_;
  const _Float16* vptr = V + (size_t)b * S_ * D_ + (size_t)h * HD_;

  const _Float16* qrow = qptr + (size_t)(q0 + ln) * D_;
  v16h aq0 = cat8(ld8(qrow + hi * 16),      ld8(qrow + hi * 16 + 8));
  v16h aq1 = cat8(ld8(qrow + 32 + hi * 16), ld8(qrow + 40 + hi * 16));

  float mrow[8], lrow[8];
#pragma unroll
  for (int r = 0; r < 8; ++r) { mrow[r] = -1e30f; lrow[r] = 0.0f; }
  v8f o0 = {}, o1 = {}, o2 = {}, o3 = {};

  for (int kc = 0; kc < S_; kc += 32) {
    int sr = tid >> 2;
    int sc = (tid & 3) * 16;
    const _Float16* krow = kptr + (size_t)(kc + sr) * D_;
    *(v8h*)&sK[sr][sc]     = ld8(krow + sc);
    *(v8h*)&sK[sr][sc + 8] = ld8(krow + sc + 8);
    const _Float16* vrow = vptr + (size_t)(kc + sr) * D_;
    v8h v0 = ld8(vrow + sc);
    v8h v1 = ld8(vrow + sc + 8);
#pragma unroll
    for (int j = 0; j < 8; ++j) { sVt[sc + j][sr] = v0[j]; sVt[sc + 8 + j][sr] = v1[j]; }
    __syncthreads();

    v8f s0 = {}, s1 = {};
    v16h bk;
    bk = cat8(ld8(&sK[ln][hi * 16]),           ld8(&sK[ln][hi * 16 + 8]));
    s0 = wmma_f16(aq0, bk, s0);
    bk = cat8(ld8(&sK[ln][32 + hi * 16]),      ld8(&sK[ln][32 + hi * 16 + 8]));
    s0 = wmma_f16(aq1, bk, s0);
    bk = cat8(ld8(&sK[16 + ln][hi * 16]),      ld8(&sK[16 + ln][hi * 16 + 8]));
    s1 = wmma_f16(aq0, bk, s1);
    bk = cat8(ld8(&sK[16 + ln][32 + hi * 16]), ld8(&sK[16 + ln][32 + hi * 16 + 8]));
    s1 = wmma_f16(aq1, bk, s1);

#pragma unroll
    for (int r = 0; r < 8; ++r) {
      int qg = q0 + r + 8 * hi;
      int kg0 = kc + ln, kg1 = kc + 16 + ln;
      float x0 = s0[r] * scale * __expf(-0.01f * fabsf((float)(qg - kg0)));
      float x1 = s1[r] * scale * __expf(-0.01f * fabsf((float)(qg - kg1)));
      float mc = fmaxf(x0, x1);
      mc = fmaxf(mc, __shfl_xor(mc, 1, 32));
      mc = fmaxf(mc, __shfl_xor(mc, 2, 32));
      mc = fmaxf(mc, __shfl_xor(mc, 4, 32));
      mc = fmaxf(mc, __shfl_xor(mc, 8, 32));
      float mnew = fmaxf(mrow[r], mc);
      float alpha = __expf(mrow[r] - mnew);
      mrow[r] = mnew;
      float p0 = __expf(x0 - mnew), p1 = __expf(x1 - mnew);
      float rs = p0 + p1;
      rs += __shfl_xor(rs, 1, 32);
      rs += __shfl_xor(rs, 2, 32);
      rs += __shfl_xor(rs, 4, 32);
      rs += __shfl_xor(rs, 8, 32);
      lrow[r] = lrow[r] * alpha + rs;
      o0[r] *= alpha; o1[r] *= alpha; o2[r] *= alpha; o3[r] *= alpha;
      sP[wave][r + 8 * hi][ln]      = (_Float16)p0;
      sP[wave][r + 8 * hi][16 + ln] = (_Float16)p1;
    }

    v16h ap = cat8(ld8(&sP[wave][ln][hi * 16]), ld8(&sP[wave][ln][hi * 16 + 8]));
    v16h bv;
    bv = cat8(ld8(&sVt[ln][hi * 16]),      ld8(&sVt[ln][hi * 16 + 8]));      o0 = wmma_f16(ap, bv, o0);
    bv = cat8(ld8(&sVt[16 + ln][hi * 16]), ld8(&sVt[16 + ln][hi * 16 + 8])); o1 = wmma_f16(ap, bv, o1);
    bv = cat8(ld8(&sVt[32 + ln][hi * 16]), ld8(&sVt[32 + ln][hi * 16 + 8])); o2 = wmma_f16(ap, bv, o2);
    bv = cat8(ld8(&sVt[48 + ln][hi * 16]), ld8(&sVt[48 + ln][hi * 16 + 8])); o3 = wmma_f16(ap, bv, o3);
    __syncthreads();
  }

  _Float16* optr = O + (size_t)b * S_ * D_ + (size_t)h * HD_;
#pragma unroll
  for (int r = 0; r < 8; ++r) {
    float inv = 1.0f / lrow[r];
    size_t row = (size_t)(q0 + r + 8 * hi);
    optr[row * D_ + ln]      = (_Float16)(o0[r] * inv);
    optr[row * D_ + 16 + ln] = (_Float16)(o1[r] * inv);
    optr[row * D_ + 32 + ln] = (_Float16)(o2[r] * inv);
    optr[row * D_ + 48 + ln] = (_Float16)(o3[r] * inv);
  }
}

// ---------------------------------------------------------------- launch
extern "C" void kernel_launch(void* const* d_in, const int* in_sizes, int n_in,
                              void* d_out, int out_size, void* d_ws, size_t ws_size,
                              hipStream_t stream) {
  (void)in_sizes; (void)n_in; (void)out_size; (void)ws_size;
  const int*   src  = (const int*)  d_in[0];
  const float* emb  = (const float*)d_in[2];
  const float* Wq   = (const float*)d_in[3];
  const float* Wk   = (const float*)d_in[4];
  const float* Wv   = (const float*)d_in[5];
  const float* Wo   = (const float*)d_in[6];
  const float* Wdn  = (const float*)d_in[7];
  const float* Wupk = (const float*)d_in[8];
  const float* Wupv = (const float*)d_in[9];
  const float* fcw  = (const float*)d_in[10];
  const float* fcb  = (const float*)d_in[11];
  float* out = (float*)d_out;

  char* ws = (char*)d_ws;
  size_t off = 0;
  auto take = [&](size_t halves) -> _Float16* {
    _Float16* p = (_Float16*)(ws + off);
    off += halves * sizeof(_Float16);
    off = (off + 255) & ~(size_t)255;
    return p;
  };
  // all weights stored TRANSPOSED: [N][K] f16
  _Float16* wq    = take((size_t)L_ * D_ * D_);
  _Float16* wk    = take((size_t)L_ * D_ * D_);
  _Float16* wv    = take((size_t)L_ * D_ * D_);
  _Float16* wo    = take((size_t)L_ * D_ * D_);
  _Float16* wdn   = take((size_t)L_ * D_ * DL_);
  _Float16* wupk  = take((size_t)L_ * DL_ * D_);
  _Float16* wupv  = take((size_t)L_ * DL_ * D_);
  _Float16* wfc   = take((size_t)D_ * V_);
  _Float16* xa    = take((size_t)M_ * D_);
  _Float16* qb    = take((size_t)M_ * D_);
  _Float16* kb    = take((size_t)M_ * D_);
  _Float16* vb    = take((size_t)M_ * D_);
  _Float16* latb  = take((size_t)M_ * DL_);
  _Float16* attnb = take((size_t)M_ * D_);

  // convert + transpose weights (K is a power of two for every matrix)
  cvtT<<<dim3((D_ * D_) / 256, L_), 256, 0, stream>>>(Wq,   wq,   10, D_ * D_);
  cvtT<<<dim3((D_ * D_) / 256, L_), 256, 0, stream>>>(Wk,   wk,   10, D_ * D_);
  cvtT<<<dim3((D_ * D_) / 256, L_), 256, 0, stream>>>(Wv,   wv,   10, D_ * D_);
  cvtT<<<dim3((D_ * D_) / 256, L_), 256, 0, stream>>>(Wo,   wo,   10, D_ * D_);
  cvtT<<<dim3((D_ * DL_) / 256, L_), 256, 0, stream>>>(Wdn,  wdn,  10, D_ * DL_);
  cvtT<<<dim3((DL_ * D_) / 256, L_), 256, 0, stream>>>(Wupk, wupk, 9,  DL_ * D_);
  cvtT<<<dim3((DL_ * D_) / 256, L_), 256, 0, stream>>>(Wupv, wupv, 9,  DL_ * D_);
  cvtT<<<dim3((D_ * V_) / 256, 1), 256, 0, stream>>>(fcw,  wfc,  10, D_ * V_);

  embed_pe<<<(M_ * D_) / 256, 256, 0, stream>>>(src, emb, xa);

  dim3 gD(D_ / 128, M_ / 128);    // (8, 32)
  dim3 gDn(DL_ / 128, M_ / 128);  // (4, 32)
  dim3 gV(V_ / 128, M_ / 128);    // (250, 32)
  dim3 gAttn(S_ / 64, B_ * H_);   // (16, 64)

  for (int l = 0; l < L_; ++l) {
    size_t wOff = (size_t)l * D_ * D_;
    size_t dOff = (size_t)l * D_ * DL_;
    size_t uOff = (size_t)l * DL_ * D_;
    gemm2<false, false><<<gD, 256, 0, stream>>>(xa, nullptr, wq + wOff, nullptr, qb, nullptr, M_, D_, D_);
    gemm2<false, false><<<gD, 256, 0, stream>>>(xa, nullptr, wk + wOff, nullptr, kb, nullptr, M_, D_, D_);
    gemm2<false, false><<<gD, 256, 0, stream>>>(xa, nullptr, wv + wOff, nullptr, vb, nullptr, M_, D_, D_);
    // latent = (k + v) @ Wdown  (fused elementwise add in A staging)
    gemm2<true, false><<<gDn, 256, 0, stream>>>(kb, vb, wdn + dOff, nullptr, latb, nullptr, M_, DL_, D_);
    // kr / vr overwrite kb / vb
    gemm2<false, false><<<gD, 256, 0, stream>>>(latb, nullptr, wupk + uOff, nullptr, kb, nullptr, M_, D_, DL_);
    gemm2<false, false><<<gD, 256, 0, stream>>>(latb, nullptr, wupv + uOff, nullptr, vb, nullptr, M_, D_, DL_);
    attn_wmma<<<gAttn, 128, 0, stream>>>(qb, kb, vb, attnb, 0.03125f /* 1/sqrt(1024) */);
    gemm2<false, false><<<gD, 256, 0, stream>>>(attnb, nullptr, wo + wOff, nullptr, xa, nullptr, M_, D_, D_);
  }
  // logits = x @ fc_w + fc_b  (f32 output)
  gemm2<false, true><<<gV, 256, 0, stream>>>(xa, nullptr, wfc, fcb, nullptr, out, M_, V_, D_);
}
